// ResidualSSM_15547781611904
// MI455X (gfx1250) — matrix operand order repaired and checked by reading, not compiled, not run on hardware
//
#include <hip/hip_runtime.h>

// ---------------------------------------------------------------------------
// Problem constants (from reference)
// ---------------------------------------------------------------------------
#define B_    8
#define L_    4096
#define D_    1024
#define MAXW  720
#define NCH   23          // ceil((720+16)/32) 32-tap chunks covering all taps
#define TS    1024        // output timesteps per block (MA kernel)
#define XT    1760        // reversed x slab length per channel (mult of 16)
#define NTILE 4           // 32-step output tiles per wave (4 WMMA acc chains)

typedef __attribute__((ext_vector_type(16))) __bf16 v16bf;
typedef __attribute__((ext_vector_type(8)))  float  v8f;

__device__ __forceinline__ __bf16 to_bf16(float f) { return (__bf16)f; }

// ---------------------------------------------------------------------------
// Kernel 1: diff channels (c<4). 4-tap signed-Pascal causal FIR. Memory bound.
// One thread per output element, fully coalesced over d.
// ---------------------------------------------------------------------------
__global__ __launch_bounds__(256)
void diff_fir(const float* __restrict__ u, const float* __restrict__ dk,
              float* __restrict__ out) {
    const long long gtid = (long long)blockIdx.x * 256 + threadIdx.x;
    const int dIdx = (int)(gtid & 511);          // 512 diff channels
    const long long rem = gtid >> 9;
    const int t = (int)(rem & (L_ - 1));
    const int b = (int)(rem >> 12);
    const int sub = dIdx & 7;                    // head-dim repeat
    const int g2  = dIdx >> 3;
    const int c   = g2 & 3;                      // pascal row
    const int r   = g2 >> 2;                     // kernel repeat
    const int d   = (r * 8 + c) * 8 + sub;       // global channel

    const float k0 = dk[c * 4 + 0];
    const float k1 = dk[c * 4 + 1];
    const float k2 = dk[c * 4 + 2];
    const float k3 = dk[c * 4 + 3];

    const size_t base = ((size_t)b * L_ + t) * D_ + d;
    float y = k0 * u[base];
    if (t >= 1) y += k1 * u[base - (size_t)D_];
    if (t >= 2) y += k2 * u[base - (size_t)(2 * D_)];
    if (t >= 3) y += k3 * u[base - (size_t)(3 * D_)];
    out[base] = y;
}

// ---------------------------------------------------------------------------
// Kernel 2: MA channels (c>=4). 720-tap causal FIR as Toeplitz matmul on
// v_wmma_f32_16x16x32_bf16.
//
// Block = (batch b, ma-kernel ki, 1024-step span). 256 threads = 8 waves.
// Wave w produces outputs y[spanBase + 128w .. +128w+127] for the 8 channels
// sharing kernel ki, as NTILE=4 independent 16x16 f32 accumulator chains
// (M=16 output steps, N = 8 channels x 2 sixteen-step blocks each).
//
// Per 32-tap chunk c:
//   A[m,k]      = h[32c-16 + k + m]          (Toeplitz band, zero-padded h)
//   B[k,(s,ch)] = x[t0 + 16s + 16 - 32c - k] (sliding window, time-reversed)
//   C[m,(s,ch)] += sum_k A B  ==> partial y[t0+16s+m, ch] over taps
//                 [32c-16+m, 32c+15+m]; chunks c=0..22 tile [0,735] exactly.
//
// A is pre-packed in LDS in WMMA lane layout, contiguous K order (16-bit A 16x32:
// e -> K=16*half+e, same order as B); x is stored time-reversed so
// every A/B operand is two aligned ds_load_b128. One A load feeds 4 WMMAs.
// ---------------------------------------------------------------------------
__global__ __launch_bounds__(256)
void ma_fir_wmma(const float* __restrict__ u, const float* __restrict__ mak,
                 float* __restrict__ out) {
    __shared__ __attribute__((aligned(32))) __bf16 Apk[NCH * 32 * 16]; // 23 KB
    __shared__ __attribute__((aligned(32))) __bf16 xrev[8 * XT];       // 27.5 KB

    const int bx   = blockIdx.x;
    const int span = bx & 3;                 // L / TS = 4 spans
    const int ki   = (bx >> 2) & 63;         // 64 MA kernels
    const int b    = bx >> 8;                // batch
    const int r    = ki >> 2;
    const int cc   = (ki & 3) + 4;
    const int chBase   = (r * 8 + cc) * 8;   // 8 channels share kernel ki
    const int spanBase = span * TS;
    const int tMax     = spanBase + TS;      // newest time staged (reversed p=0)
    const int tid = threadIdx.x;

    // ---- stage A: WMMA-ready Toeplitz tiles of the kernel taps -------------
    const float* hrow = mak + ki * MAXW;
    for (int idx = tid; idx < NCH * 32 * 16; idx += 256) {
        const int e  = idx & 15;
        const int ln = (idx >> 4) & 31;
        const int c  = idx >> 9;
        const int hf = ln >> 4;
        const int mm = ln & 15;
        // element e holds K = 16*hf+e; tap j = 32c-16+K+mm
        const int j = 32 * c - 16 + 16 * hf + mm + e;
        const float v = (j >= 0 && j < MAXW) ? hrow[j] : 0.0f;
        Apk[idx] = to_bf16(v);
    }

    // ---- stage x, time-reversed: xrev[ch][p] = u[b, tMax - p, chBase+ch] ---
    for (int idx = tid; idx < 8 * XT; idx += 256) {
        const int ch = idx & 7;
        const int p  = idx >> 3;
        const int t  = tMax - p;
        float v = 0.0f;
        if (t >= 0 && t < L_)
            v = u[((size_t)b * L_ + t) * D_ + (chBase + ch)];
        xrev[ch * XT + p] = to_bf16(v);
    }
    __syncthreads();

    const int lane  = tid & 31;
    const int wv    = tid >> 5;
    const int half  = lane >> 4;       // lanes 16-31 = second M/K half
    const int nCol  = lane & 15;       // B/D column
    const int sCol  = nCol >> 3;       // 16-step sub-block
    const int chCol = nCol & 7;        // channel within group
    const __bf16* xch = xrev + chCol * XT;

    // reversed-index base of the B window for tile 0 (tiles step by -32);
    // all values are multiples of 16 -> 32B-aligned v16bf loads
    const int pb0 = TS - wv * 128 - 16 * sCol - 16 + 16 * half;

    v8f acc[NTILE];
#pragma unroll
    for (int tile = 0; tile < NTILE; ++tile)
        acc[tile] = (v8f){0.f, 0.f, 0.f, 0.f, 0.f, 0.f, 0.f, 0.f};

    for (int c = 0; c < NCH; ++c) {
        const v16bf a = *(const v16bf*)(Apk + (c * 32 + lane) * 16);
        const __bf16* bbase = xch + (pb0 + 32 * c);
#pragma unroll
        for (int tile = 0; tile < NTILE; ++tile) {
            const v16bf bm = *(const v16bf*)(bbase - 32 * tile);
            acc[tile] = __builtin_amdgcn_wmma_f32_16x16x32_bf16(
                            false, a, false, bm, (short)0, acc[tile],
                            false, false);
        }
    }

    // ---- write D: VGPR i -> row M = i + 8*half, col = (sCol,chCol) ---------
    const int d = chBase + chCol;
    const int tB0 = spanBase + wv * 128 + 16 * sCol + 8 * half; // tile 0
#pragma unroll
    for (int tile = 0; tile < NTILE; ++tile) {
        const size_t rowBase = (size_t)b * L_ + (tB0 + 32 * tile);
#pragma unroll
        for (int i = 0; i < 8; ++i)
            out[(rowBase + i) * D_ + d] = acc[tile][i];
    }
}

// ---------------------------------------------------------------------------
// Launch
// ---------------------------------------------------------------------------
extern "C" void kernel_launch(void* const* d_in, const int* in_sizes, int n_in,
                              void* d_out, int out_size, void* d_ws, size_t ws_size,
                              hipStream_t stream) {
    const float* u   = (const float*)d_in[0];   // (B, L, D) f32
    const float* dk  = (const float*)d_in[1];   // (64, 4)   f32
    const float* mak = (const float*)d_in[2];   // (64, 720) f32
    float* out = (float*)d_out;                 // (B, L, D) f32

    // diff channels: one thread per output, 512 channels
    const int n_diff_blocks = (B_ * L_ * 512) / 256;       // 65536
    diff_fir<<<n_diff_blocks, 256, 0, stream>>>(u, dk, out);

    // MA channels: (batch, kernel, span) blocks
    const int n_ma_blocks = B_ * 64 * (L_ / TS);           // 2048
    ma_fir_wmma<<<n_ma_blocks, 256, 0, stream>>>(u, mak, out);

    (void)in_sizes; (void)n_in; (void)out_size; (void)d_ws; (void)ws_size;
}
